// CausalSelfAttention_18519898980510
// MI455X (gfx1250) — matrix operand order repaired and checked by reading, not compiled, not run on hardware
//
#include <hip/hip_runtime.h>
#include <hip/hip_bf16.h>

typedef __attribute__((ext_vector_type(16))) __bf16 bf16x16;
typedef __attribute__((ext_vector_type(8)))  float  f32x8;
typedef __attribute__((ext_vector_type(4)))  unsigned int u32x4;
typedef __attribute__((ext_vector_type(8)))  int   i32x8;
typedef __attribute__((ext_vector_type(4)))  int   i32x4;

#define NBATCH 4
#define SEQ    4096
#define DMODEL 256
#define NH     4
#define HDIM   64
#define ATT_SCALE 0.125f        // 1/sqrt(64)
#define LOG2E  1.4426950408889634f

struct alignas(16) U128 { unsigned int w[4]; };
union FragU { bf16x16 v; U128 q[2]; };

// ---- fragment loaders (CDNA5 wave32 WMMA layouts, 16-bit operands) ----
// A-matrix 16x32 (MxK): lane m(=lane&15) holds K=kb..kb+7 (elems 0-7) and
// K=kb+16..kb+23 (elems 8-15), kb = 8*(lane>=16). Row-major [M][K] source.
__device__ inline bf16x16 load_a_frag(const __bf16* base, int ld) {
  int lane = threadIdx.x & 31;
  int m  = lane & 15;
  int kb = (lane >> 4) << 3;
  const __bf16* p = base + m * ld + kb;
  FragU u;
  u.q[0] = *(const U128*)(p);
  u.q[1] = *(const U128*)(p + 16);
  return u.v;
}

// B-matrix 32x16 (KxN): lane n(=lane&15) holds K=kb..kb+7 (elems 0-7) and
// K=kb+16..kb+23 (elems 8-15), kb = 8*(lane>=16). Source stored row-major as [N][K] (i.e. B-transposed).
__device__ inline bf16x16 load_b_frag(const __bf16* base, int ld) {
  int lane = threadIdx.x & 31;
  int n  = lane & 15;
  int kb = (lane >> 4) << 3;
  const __bf16* p = base + n * ld + kb;
  FragU u;
  u.q[0] = *(const U128*)(p);
  u.q[1] = *(const U128*)(p + 16);
  return u.v;
}

__device__ inline f32x8 wmma_bf16(bf16x16 a, bf16x16 b, f32x8 c) {
  return __builtin_amdgcn_wmma_f32_16x16x32_bf16(
      /*neg_a=*/false, a, /*neg_b=*/false, b,
      /*c_mod=*/(short)0, c, /*reuse_a=*/false, /*reuse_b=*/false);
}

// ---- Tensor Data Mover: 2-D tile (bf16) global -> LDS ----
// D# layout per CDNA5 ISA (groups 2/3 zero => 2-D).
// This toolchain exposes the 6-arg builtin form.
__device__ inline void tdm_load_2d_bf16(unsigned lds_off, const void* gptr,
                                        unsigned tensor_d0, unsigned tensor_d1,
                                        unsigned tile_d0, unsigned tile_d1,
                                        unsigned stride_d0) {
  unsigned long long ga = (unsigned long long)(uintptr_t)gptr;
  u32x4 g0;
  g0[0] = 1u;                                            // count=1 (valid D#)
  g0[1] = lds_off;                                       // lds_addr (bytes)
  g0[2] = (unsigned)(ga & 0xFFFFFFFFu);                  // global_addr[31:0]
  g0[3] = (unsigned)((ga >> 32) & 0x01FFFFFFu) | (2u << 30); // [56:32] | type=2
  i32x8 g1;
  g1[0] = (int)(1u << 16);                               // data_size=1 (2 bytes)
  g1[1] = (int)((tensor_d0 & 0xFFFFu) << 16);            // tensor_dim0[15:0]
  g1[2] = (int)(((tensor_d0 >> 16) & 0xFFFFu) |
                ((tensor_d1 & 0xFFFFu) << 16));          // dim0 hi | dim1 lo
  g1[3] = (int)(((tensor_d1 >> 16) & 0xFFFFu) |
                ((tile_d0 & 0xFFFFu) << 16));            // dim1 hi | tile_dim0
  g1[4] = (int)(tile_d1 & 0xFFFFu);                      // tile_dim1, tile_dim2=0
  g1[5] = (int)stride_d0;                                // dim0_stride[31:0]
  g1[6] = 0;                                             // stride hi, dim1_stride lo
  g1[7] = 0;
  i32x4 z4 = {};
  i32x8 z8 = {};
  __builtin_amdgcn_tensor_load_to_lds(g0, g1, z4, z4, z8, 0);
}

// ---------------- kernel 0: fp32 -> bf16 conversion ----------------
__global__ void cvt_bf16_kernel(const float* __restrict__ x,
                                const float* __restrict__ wq, const float* __restrict__ wk,
                                const float* __restrict__ wv, const float* __restrict__ wo,
                                __bf16* __restrict__ xb,
                                __bf16* __restrict__ wqb, __bf16* __restrict__ wkb,
                                __bf16* __restrict__ wvb, __bf16* __restrict__ wob) {
  const int NX = NBATCH * SEQ * DMODEL;
  const int NW = DMODEL * DMODEL;
  const int total = NX + 4 * NW;
  for (int idx = blockIdx.x * blockDim.x + threadIdx.x; idx < total;
       idx += gridDim.x * blockDim.x) {
    if (idx < NX) {
      xb[idx] = (__bf16)x[idx];
    } else {
      int r = idx - NX;
      int w = r / NW, off = r % NW;
      const float* s = (w == 0) ? wq : (w == 1) ? wk : (w == 2) ? wv : wo;
      __bf16* d      = (w == 0) ? wqb : (w == 1) ? wkb : (w == 2) ? wvb : wob;
      d[off] = (__bf16)s[off];
    }
  }
}

// ---------------- kernel 1: QKV projection ----------------
__global__ __launch_bounds__(256) void qkv_kernel(
    const __bf16* __restrict__ xb,
    const __bf16* __restrict__ wqb, const __bf16* __restrict__ wkb,
    const __bf16* __restrict__ wvb,
    __bf16* __restrict__ qb, __bf16* __restrict__ kb, __bf16* __restrict__ vtb) {
  int wave = threadIdx.x >> 5;
  int lane = threadIdx.x & 31;
  int rowTile = blockIdx.x * 8 + wave;      // 0..1023
  int rowbase = rowTile * 16;               // global row in [0,16384)
  int b  = rowbase >> 12;                   // row / 4096
  int s0 = rowbase & 4095;
  const __bf16* arow = xb + (long long)rowbase * DMODEL;
  int half = lane >> 4, col0 = lane & 15;

  for (int h = 0; h < NH; ++h) {
    f32x8 accq[4] = {}, acck[4] = {}, accv[4] = {};
    for (int kk = 0; kk < 8; ++kk) {          // K-dim 256 = 8 x 32
      bf16x16 a = load_a_frag(arow + kk * 32, DMODEL);
#pragma unroll
      for (int ni = 0; ni < 4; ++ni) {
        int orow = h * 64 + ni * 16;
        accq[ni] = wmma_bf16(a, load_b_frag(wqb + orow * DMODEL + kk * 32, DMODEL), accq[ni]);
        acck[ni] = wmma_bf16(a, load_b_frag(wkb + orow * DMODEL + kk * 32, DMODEL), acck[ni]);
        accv[ni] = wmma_bf16(a, load_b_frag(wvb + orow * DMODEL + kk * 32, DMODEL), accv[ni]);
      }
    }
    __bf16* qh = qb  + (long long)(b * NH + h) * SEQ * HDIM;
    __bf16* kh = kb  + (long long)(b * NH + h) * SEQ * HDIM;
    __bf16* vh = vtb + (long long)(b * NH + h) * HDIM * SEQ;
#pragma unroll
    for (int ni = 0; ni < 4; ++ni)
#pragma unroll
      for (int i = 0; i < 8; ++i) {
        int srow = s0 + i + half * 8;          // C frag: row = i + 8*(lane>=16)
        int col  = ni * 16 + col0;             //         col = lane&15 (+16*ni)
        qh[(long long)srow * HDIM + col] = (__bf16)accq[ni][i];
        kh[(long long)srow * HDIM + col] = (__bf16)acck[ni][i];
        vh[(long long)col * SEQ + srow]  = (__bf16)accv[ni][i];
      }
  }
}

// ---------------- kernel 2: causal flash attention ----------------
// All 8 waves of a block share (b,h); K/V chunks are TDM-staged into LDS
// once per block with double buffering; each wave owns 16 query rows and
// does its own online softmax inside 16-lane halves.
__global__ __launch_bounds__(256) void attn_kernel(
    const __bf16* __restrict__ qb, const __bf16* __restrict__ kb,
    const __bf16* __restrict__ vtb, __bf16* __restrict__ ob) {
  __shared__ __attribute__((aligned(16))) __bf16 kstage[2][64 * 64];  // [key][hd]
  __shared__ __attribute__((aligned(16))) __bf16 vstage[2][64 * 64];  // [hd][key]
  __shared__ __attribute__((aligned(16))) __bf16 pstage[8][16 * 80];  // per-wave P

  int wave = threadIdx.x >> 5;
  int lane = threadIdx.x & 31;
  int task = blockIdx.x * 8 + wave;           // 0..4095
  int qt = task & 255;
  int bh = task >> 8;
  int b = bh >> 2, h = bh & 3;
  int qbase = qt * 16;
  int half = lane >> 4, col0 = lane & 15;

  const __bf16* qh = qb  + (long long)(b * NH + h) * SEQ * HDIM;
  const __bf16* kh = kb  + (long long)(b * NH + h) * SEQ * HDIM;
  const __bf16* vh = vtb + (long long)(b * NH + h) * HDIM * SEQ;

  bf16x16 aq0 = load_a_frag(qh + (long long)qbase * HDIM, HDIM);
  bf16x16 aq1 = load_a_frag(qh + (long long)qbase * HDIM + 32, HDIM);

  f32x8 o[4] = {};
  float rmax[8], rsum[8];
#pragma unroll
  for (int i = 0; i < 8; ++i) { rmax[i] = -3.0e38f; rsum[i] = 0.f; }

  __bf16* pst = &pstage[wave][0];
  // block-wide chunk count: last wave in block has the largest causal extent
  int qbaseLast = (((blockIdx.x * 8) & 255) + 7) * 16;
  int nChunks = (qbaseLast + 15) / 64 + 1;

  // prologue: TDM-stage chunk 0
  if (wave == 0) {
    tdm_load_2d_bf16((unsigned)(uintptr_t)&kstage[0][0], kh,
                     HDIM, SEQ, HDIM, 64, HDIM);
    tdm_load_2d_bf16((unsigned)(uintptr_t)&vstage[0][0], vh,
                     SEQ, HDIM, 64, HDIM, SEQ);
  }

  for (int j = 0; j < nChunks; ++j) {
    int jb = j * 64;
    if (wave == 0) {
      if (j + 1 < nChunks) {   // issue next chunk, then wait for current
        int nb = jb + 64;
        tdm_load_2d_bf16((unsigned)(uintptr_t)&kstage[(j + 1) & 1][0],
                         kh + (long long)nb * HDIM, HDIM, SEQ, HDIM, 64, HDIM);
        tdm_load_2d_bf16((unsigned)(uintptr_t)&vstage[(j + 1) & 1][0],
                         vh + nb, SEQ, HDIM, 64, HDIM, SEQ);
        __builtin_amdgcn_s_wait_tensorcnt(2);  // 2 outstanding = next K+V only
      } else {
        __builtin_amdgcn_s_wait_tensorcnt(0);
      }
    }
    __syncthreads();                           // chunk j resident in LDS
    const __bf16* kst = &kstage[j & 1][0];
    const __bf16* vst = &vstage[j & 1][0];

    if (jb <= qbase + 15) {                    // causal: this wave needs chunk j
      // ---- S = Q K^T ----
      f32x8 sA[4] = {};
#pragma unroll
      for (int ni = 0; ni < 4; ++ni) {
        const __bf16* kt = kst + (ni * 16) * HDIM;
        sA[ni] = wmma_bf16(aq0, load_b_frag(kt, HDIM), sA[ni]);
        sA[ni] = wmma_bf16(aq1, load_b_frag(kt + 32, HDIM), sA[ni]);
      }
      // ---- scale + causal mask ----
#pragma unroll
      for (int ni = 0; ni < 4; ++ni) {
        int key = jb + ni * 16 + col0;
#pragma unroll
        for (int i = 0; i < 8; ++i) {
          float sv = sA[ni][i] * ATT_SCALE;
          int row = qbase + i + half * 8;
          if (key > row) sv = -3.0e38f;
          sA[ni][i] = sv;
        }
      }
      // ---- online softmax (8 rows per 16-lane half) ----
#pragma unroll
      for (int i = 0; i < 8; ++i) {
        float m = fmaxf(fmaxf(sA[0][i], sA[1][i]), fmaxf(sA[2][i], sA[3][i]));
        m = fmaxf(m, __shfl_xor(m, 1, 16));
        m = fmaxf(m, __shfl_xor(m, 2, 16));
        m = fmaxf(m, __shfl_xor(m, 4, 16));
        m = fmaxf(m, __shfl_xor(m, 8, 16));
        float mnew = fmaxf(rmax[i], m);
        float corr = exp2f((rmax[i] - mnew) * LOG2E);
        rmax[i] = mnew;
        float psum = 0.f;
#pragma unroll
        for (int ni = 0; ni < 4; ++ni) {
          float p = exp2f((sA[ni][i] - mnew) * LOG2E);
          sA[ni][i] = p;
          psum += p;
        }
        psum += __shfl_xor(psum, 1, 16);
        psum += __shfl_xor(psum, 2, 16);
        psum += __shfl_xor(psum, 4, 16);
        psum += __shfl_xor(psum, 8, 16);
        rsum[i] = rsum[i] * corr + psum;
#pragma unroll
        for (int ni = 0; ni < 4; ++ni) o[ni][i] *= corr;
      }
      // ---- restage P (C layout) -> A-frag layout via LDS ----
#pragma unroll
      for (int ni = 0; ni < 4; ++ni)
#pragma unroll
        for (int i = 0; i < 8; ++i)
          pst[(i + half * 8) * 80 + ni * 16 + col0] = (__bf16)sA[ni][i];
      asm volatile("s_wait_dscnt 0" ::: "memory");
      bf16x16 ap0 = load_a_frag(pst, 80);
      bf16x16 ap1 = load_a_frag(pst + 32, 80);
      // ---- O += P V ----
#pragma unroll
      for (int ni = 0; ni < 4; ++ni) {
        const __bf16* vt = vst + (ni * 16) * 64;
        o[ni] = wmma_bf16(ap0, load_b_frag(vt, 64), o[ni]);
        o[ni] = wmma_bf16(ap1, load_b_frag(vt + 32, 64), o[ni]);
      }
    }
    __syncthreads();   // buf[j&1] free before it is refilled at iter j+1
  }
  // ---- normalize + store to merged [B*S][256] bf16 buffer ----
#pragma unroll
  for (int ni = 0; ni < 4; ++ni)
#pragma unroll
    for (int i = 0; i < 8; ++i) {
      int srow = qbase + i + half * 8;
      float val = o[ni][i] / rsum[i];
      ob[(long long)(b * SEQ + srow) * DMODEL + h * 64 + ni * 16 + col0] = (__bf16)val;
    }
}

// ---------------- kernel 3: output projection (fp32 out) ----------------
__global__ __launch_bounds__(256) void oproj_kernel(
    const __bf16* __restrict__ ab, const __bf16* __restrict__ wob,
    float* __restrict__ out) {
  int wave = threadIdx.x >> 5;
  int lane = threadIdx.x & 31;
  int task = blockIdx.x * 8 + wave;          // 0..4095
  int rowTile = task >> 2;                   // 0..1023
  int cg = task & 3;                         // 64-col group
  int rowbase = rowTile * 16;
  const __bf16* arow = ab + (long long)rowbase * DMODEL;
  f32x8 acc[4] = {};
  for (int kk = 0; kk < 8; ++kk) {
    bf16x16 a = load_a_frag(arow + kk * 32, DMODEL);
#pragma unroll
    for (int ni = 0; ni < 4; ++ni)
      acc[ni] = wmma_bf16(a, load_b_frag(wob + (cg * 64 + ni * 16) * DMODEL + kk * 32, DMODEL),
                          acc[ni]);
  }
  int half = lane >> 4, col0 = lane & 15;
#pragma unroll
  for (int ni = 0; ni < 4; ++ni)
#pragma unroll
    for (int i = 0; i < 8; ++i)
      out[(long long)(rowbase + i + half * 8) * DMODEL + cg * 64 + ni * 16 + col0] =
          acc[ni][i];
}

extern "C" void kernel_launch(void* const* d_in, const int* in_sizes, int n_in,
                              void* d_out, int out_size, void* d_ws, size_t ws_size,
                              hipStream_t stream) {
  const float* x  = (const float*)d_in[0];
  const float* Wq = (const float*)d_in[1];
  const float* Wk = (const float*)d_in[2];
  const float* Wv = (const float*)d_in[3];
  const float* Wo = (const float*)d_in[4];
  float* out = (float*)d_out;

  char* ws = (char*)d_ws;
  const size_t szX = (size_t)NBATCH * SEQ * DMODEL * sizeof(__bf16);    // 8 MB
  const size_t szW = (size_t)DMODEL * DMODEL * sizeof(__bf16);          // 128 KB
  const size_t szH = (size_t)NBATCH * NH * SEQ * HDIM * sizeof(__bf16); // 8 MB

  __bf16* xb  = (__bf16*)(ws);
  __bf16* wqb = (__bf16*)(ws + szX);
  __bf16* wkb = (__bf16*)(ws + szX + szW);
  __bf16* wvb = (__bf16*)(ws + szX + 2 * szW);
  __bf16* wob = (__bf16*)(ws + szX + 3 * szW);
  __bf16* qb  = (__bf16*)(ws + szX + 4 * szW);
  __bf16* kb  = (__bf16*)(ws + szX + 4 * szW + szH);
  __bf16* vtb = (__bf16*)(ws + szX + 4 * szW + 2 * szH);
  __bf16* ab  = (__bf16*)(ws + szX + 4 * szW + 3 * szH);

  cvt_bf16_kernel<<<2048, 256, 0, stream>>>(x, Wq, Wk, Wv, Wo, xb, wqb, wkb, wvb, wob);
  qkv_kernel<<<128, 256, 0, stream>>>(xb, wqb, wkb, wvb, qb, kb, vtb);
  attn_kernel<<<512, 256, 0, stream>>>(qb, kb, vtb, ab);
  oproj_kernel<<<512, 256, 0, stream>>>(ab, wob, out);
}
